// CrfLayer_11690900979888
// MI455X (gfx1250) — matrix operand order repaired and checked by reading, not compiled, not run on hardware
//
#include <hip/hip_runtime.h>

// Linear-chain CRF forward partition function, T=512, B=512, K=64.
//
// Recurrence rewritten as a GEMM in probability space (base-2 on-chip):
//   new_fv[b,j] = emit[t,b,j] + m_b + c_j + ln2*log2( sum_i exp2((fv[b,i]-m_b)*log2e)
//                                                         * exp(trans[i,j]-c_j) )
// Contraction via v_wmma_f32_16x16x32_f16. E = exp(trans - c) lives in WMMA
// B-fragment registers for all 512 steps. One wave (32 lanes) per workgroup
// owns a 16-row batch tile; 32 workgroups ride the latency chain in parallel.

typedef __attribute__((ext_vector_type(16))) _Float16 v16h;
typedef __attribute__((ext_vector_type(8)))  _Float16 v8h;
typedef __attribute__((ext_vector_type(4)))  _Float16 v4h;
typedef __attribute__((ext_vector_type(8)))  float    v8f;

#define TT 512
#define BB 512
#define KK 64
#define FV_STRIDE 68   // floats per LDS row (padded, float4-aligned)
#define P_STRIDE  72   // halves per LDS row (padded, keeps 16B chunk alignment)

#define F_LOG2E 1.4426950408889634f
#define F_LN2   0.6931471805599453f

// Raw hardware transcendentals (v_exp_f32 / v_log_f32, base-2).
#if __has_builtin(__builtin_amdgcn_exp2f)
#define EXP2F(x) __builtin_amdgcn_exp2f(x)
#else
#define EXP2F(x) exp2f(x)
#endif
#if __has_builtin(__builtin_amdgcn_logf)
#define LOG2F(x) __builtin_amdgcn_logf(x)
#else
#define LOG2F(x) log2f(x)
#endif

__global__ __launch_bounds__(32)
void crf_forward_kernel(const float* __restrict__ emits,
                        const unsigned char* __restrict__ mask,
                        const float* __restrict__ trans,
                        const float* __restrict__ alpha0,
                        float* __restrict__ ws)
{
  __shared__ __align__(16) float    FV[16 * FV_STRIDE];  // fv (C layout rows)
  __shared__ __align__(16) _Float16 P [16 * P_STRIDE];   // exp(fv - m), f16
  __shared__ __align__(16) float    mLds[16];
  __shared__ float cLds[KK];

  const int lane    = threadIdx.x;   // 0..31
  const int row     = lane & 15;
  const int hi      = lane >> 4;     // lane half
  const int colBase = hi * 32;
  const int b0      = blockIdx.x * 16;

  // ---- c_j = max_i trans[i][j] (once)
  for (int j = lane; j < KK; j += 32) {
    float c = trans[j];
    for (int i = 1; i < KK; ++i) c = fmaxf(c, trans[i * KK + j]);
    cLds[j] = c;
  }
  __syncthreads();

  // ---- E fragments (WMMA B layout), in VGPRs for the whole T loop.
  // B 32x16 f16, contiguous k order used here (same as A): lane L, element h: K(i) = h + (L>=16)*16 + 32*kc,
  //                                         N(j) = (L&15) + 16*jt
  v16h eb[4][2];
  float cR[4];
#pragma unroll
  for (int jt = 0; jt < 4; ++jt) {
    const int j = jt * 16 + row;
    cR[jt] = cLds[j];
#pragma unroll
    for (int kc = 0; kc < 2; ++kc) {
#pragma unroll
      for (int h = 0; h < 16; ++h) {
        const int i = kc * 32 + hi * 16 + h;
        eb[jt][kc][h] = (_Float16)expf(trans[i * KK + j] - cR[jt]);
      }
    }
  }

  // ---- ws[b] = logsumexp_j fv[b,:] on the masked (last-valid) step
  auto do_mask = [&](int t) {
    const unsigned char mk = mask[(size_t)t * BB + b0 + row];
    if (mk) {                                  // partner lane (L^16) has same mk
      const float* fr = &FV[row * FV_STRIDE + colBase];
      float4 q[8];
#pragma unroll
      for (int v = 0; v < 8; ++v) q[v] = *(const float4*)(fr + v * 4);
      float mx = q[0].x;
#pragma unroll
      for (int v = 0; v < 8; ++v) {
        mx = fmaxf(mx, fmaxf(fmaxf(q[v].x, q[v].y), fmaxf(q[v].z, q[v].w)));
      }
      mx = fmaxf(mx, __shfl_xor(mx, 16, 32));
      float s = 0.f;
#pragma unroll
      for (int v = 0; v < 8; ++v) {
        s += expf(q[v].x - mx) + expf(q[v].y - mx)
           + expf(q[v].z - mx) + expf(q[v].w - mx);
      }
      s += __shfl_xor(s, 16, 32);
      if (hi == 0) ws[b0 + row] = mx + logf(s);
    }
  };

  // ---- t = 0: fv0 = alpha0^T + emits[0], written directly in C layout
  {
    const float* ep = emits + (size_t)b0 * KK;
#pragma unroll
    for (int jt = 0; jt < 4; ++jt) {
      const int j = jt * 16 + row;
      const float a0 = alpha0[j];
#pragma unroll
      for (int r = 0; r < 8; ++r) {
        const int M = r + hi * 8;
        FV[M * FV_STRIDE + j] = a0 + ep[M * KK + j];
      }
    }
  }
  __syncthreads();
  do_mask(0);

  for (int t = 1; t < TT; ++t) {
    if (t + 1 < TT)  // one step ahead; each lane's byte pulls one cacheline
      __builtin_prefetch(emits + ((size_t)(t + 1) * BB + b0) * KK + lane * 32, 0, 3);

    // phase 1: vector-load own half-row of fv, row max
    const float* fr = &FV[row * FV_STRIDE + colBase];
    float4 q[8];
#pragma unroll
    for (int v = 0; v < 8; ++v) q[v] = *(const float4*)(fr + v * 4);
    float mx = q[0].x;
#pragma unroll
    for (int v = 0; v < 8; ++v) {
      mx = fmaxf(mx, fmaxf(fmaxf(q[v].x, q[v].y), fmaxf(q[v].z, q[v].w)));
    }
    mx = fmaxf(mx, __shfl_xor(mx, 16, 32));
    if (hi == 0) mLds[row] = mx;

    // phase 2: P = exp2(fv*log2e - m*log2e) as f16 -> v_fma + v_exp_f32 each
    const float nml2 = -mx * F_LOG2E;
    _Float16* pw = &P[row * P_STRIDE + colBase];
#pragma unroll
    for (int v = 0; v < 8; ++v) {
      v4h pk;
      pk[0] = (_Float16)EXP2F(fmaf(q[v].x, F_LOG2E, nml2));
      pk[1] = (_Float16)EXP2F(fmaf(q[v].y, F_LOG2E, nml2));
      pk[2] = (_Float16)EXP2F(fmaf(q[v].z, F_LOG2E, nml2));
      pk[3] = (_Float16)EXP2F(fmaf(q[v].w, F_LOG2E, nml2));
      *(v4h*)(pw + v * 4) = pk;
    }
    __syncthreads();                           // B1: old fv consumed, P/m published

    // row maxes into registers (two broadcast float4 loads)
    const float4 m01 = *(const float4*)&mLds[hi * 8];
    const float4 m23 = *(const float4*)&mLds[hi * 8 + 4];
    const float mreg[8] = { m01.x, m01.y, m01.z, m01.w,
                            m23.x, m23.y, m23.z, m23.w };

    // A fragments from P.  A 16x32 f16, contiguous k order used here (same as B): lane L, element e: M = L&15,
    //   K = e + (L>=16)*16 + 32*kc  -> two 16B LDS loads each
    v16h a[2];
#pragma unroll
    for (int kc = 0; kc < 2; ++kc) {
      const int kb = kc * 32 + hi * 16;
      const _Float16* pp = &P[row * P_STRIDE];
      const v8h lo  = *(const v8h*)(pp + kb);
      const v8h hih = *(const v8h*)(pp + kb + 8);
#pragma unroll
      for (int h = 0; h < 8; ++h) { a[kc][h] = lo[h]; a[kc][8 + h] = hih[h]; }
    }

    // emits[t] loaded straight into C layout (64B-coalesced per (jt,r) group)
    const float* ep = emits + ((size_t)t * BB + b0) * KK;
    float em[4][8];
#pragma unroll
    for (int jt = 0; jt < 4; ++jt) {
      const int j = jt * 16 + row;
#pragma unroll
      for (int r = 0; r < 8; ++r) em[jt][r] = ep[(r + hi * 8) * KK + j];
    }

    // phase 3: S = P @ E via WMMA; fv = (m+c+emit) + ln2*log2(S)
    // S >= exp(min(trans)-max(trans)) ~ 6e-3: never denormal, raw v_log_f32 ok.
#pragma unroll
    for (int jt = 0; jt < 4; ++jt) {
      v8f acc = {};
      acc = __builtin_amdgcn_wmma_f32_16x16x32_f16(false, a[0], false, eb[jt][0],
                                                   (short)0, acc, false, false);
      acc = __builtin_amdgcn_wmma_f32_16x16x32_f16(false, a[1], false, eb[jt][1],
                                                   (short)0, acc, false, false);
      const int j = jt * 16 + row;
#pragma unroll
      for (int r = 0; r < 8; ++r) {
        const float mc = mreg[r] + cR[jt] + em[jt][r];
        const float nv = fmaf(F_LN2, LOG2F(acc[r]), mc);
        FV[(r + hi * 8) * FV_STRIDE + j] = nv;
      }
    }
    __syncthreads();                           // B2: new fv published

    do_mask(t);
  }
}

// Deterministic fixed-order reduction of the 512 per-batch logsumexp values.
__global__ __launch_bounds__(32)
void crf_reduce_kernel(const float* __restrict__ ws, float* __restrict__ out)
{
  if (threadIdx.x == 0) {
    float s = 0.f;
    for (int b = 0; b < BB; ++b) s += ws[b];
    out[0] = s;
  }
}

extern "C" void kernel_launch(void* const* d_in, const int* in_sizes, int n_in,
                              void* d_out, int out_size, void* d_ws, size_t ws_size,
                              hipStream_t stream)
{
  const float*         emits  = (const float*)d_in[0];
  const unsigned char* mask   = (const unsigned char*)d_in[1]; // jax bool = 1 byte
  const float*         trans  = (const float*)d_in[2];
  const float*         alpha0 = (const float*)d_in[3];
  float* ws  = (float*)d_ws;   // 512 floats; every slot written exactly once
  float* out = (float*)d_out;
  (void)in_sizes; (void)n_in; (void)out_size; (void)ws_size;

  crf_forward_kernel<<<dim3(BB / 16), dim3(32), 0, stream>>>(emits, mask, trans, alpha0, ws);
  crf_reduce_kernel<<<dim3(1), dim3(32), 0, stream>>>(ws, out);
}
